// MambaLayer_3599182594615
// MI455X (gfx1250) — matrix operand order repaired and checked by reading, not compiled, not run on hardware
//
#include <hip/hip_runtime.h>
#include <hip/hip_bf16.h>
#include <math.h>

typedef __attribute__((ext_vector_type(16))) _Float16 v16h;
typedef __attribute__((ext_vector_type(8)))  float    v8f;

#define DMODEL 128
#define DINNER 256
#define DSTATE 16
#define DTRANK 8
#define NXPAD  48                /* DT_RANK + 2*D_STATE = 40, padded to 48 */
#define LSEQ   8192
#define BSZ    2
#define NTOK   (BSZ * LSEQ)      /* 16384 */
#define NCHUNK 64                /* chunks per batch for parallel scan */
#define CHLEN  (LSEQ / NCHUNK)   /* 128 steps per chunk */

// ---------------------------------------------------------------------------
// CDNA5 async global->LDS staging (ASYNCcnt-tracked)
// ---------------------------------------------------------------------------
static __device__ __forceinline__ void async_lds_b128(unsigned lds_off,
                                                      const void* gaddr) {
  asm volatile("global_load_async_to_lds_b128 %0, %1, off"
               :: "v"(lds_off), "v"(gaddr) : "memory");
}
static __device__ __forceinline__ void async_wait0() {
  asm volatile("s_wait_asynccnt 0x0" ::: "memory");
}

// ---------------------------------------------------------------------------
// WMMA fragment loaders (CDNA5 wave32 layouts, 16-bit operands, K=32 tiles)
// ---------------------------------------------------------------------------
static __device__ __forceinline__ v16h load_fragA(const _Float16* base, int ld,
                                                  int row0, int k0) {
  const int lane = threadIdx.x & 31;
  const int m = lane & 15, hi = lane >> 4;
  // A 16x32 f16: VGPR0-3 hold K = hi*8 + 0..7 ; VGPR4-7 hold K = 16 + hi*8 + 0..7
  const _Float16* p = base + (size_t)(row0 + m) * ld + k0 + hi * 8;
  v16h f;
  ((float4*)&f)[0] = *(const float4*)(p);
  ((float4*)&f)[1] = *(const float4*)(p + 16);
  return f;
}

// A fragment from an LDS-resident 16 x ld tile (dynamic LDS -> real ds loads)
static __device__ __forceinline__ v16h load_fragA_sh(const _Float16* sh, int ld,
                                                     int k0) {
  const int lane = threadIdx.x & 31;
  const int m = lane & 15, hi = lane >> 4;
  const _Float16* p = sh + m * ld + k0 + hi * 8;
  v16h f;
  ((float4*)&f)[0] = *(const float4*)(p);
  ((float4*)&f)[1] = *(const float4*)(p + 16);
  return f;
}

static __device__ __forceinline__ v16h load_fragB(const _Float16* w, int ld,
                                                  int n0, int k0) {
  const int lane = threadIdx.x & 31;
  const int n = lane & 15, hi = lane >> 4;
  // B 32x16 f16: VGPR0-3 hold K = hi*8 + 0..7 ; VGPR4-7 hold K = 16 + hi*8 + 0..7
  const _Float16* p = w + (size_t)(n0 + n) * ld + k0 + hi * 8;
  v16h f;
  ((float4*)&f)[0] = *(const float4*)(p);
  ((float4*)&f)[1] = *(const float4*)(p + 16);
  return f;
}

static __device__ __forceinline__ float silu_f(float v) {
  return v / (1.0f + __expf(-v));
}

// ---------------------------------------------------------------------------
// K0: convert weights to f16: W_in 512x128, W_out 128x256, W_x padded 48x256
// ---------------------------------------------------------------------------
__global__ void k_cvt_weights(const float* W_in, const float* W_out,
                              const float* W_x, _Float16* WinH,
                              _Float16* WoutH, _Float16* WxH) {
  int i = blockIdx.x * blockDim.x + threadIdx.x;
  if (i < 512 * 128) WinH[i] = (_Float16)W_in[i];
  if (i < 128 * 256) WoutH[i] = (_Float16)W_out[i];
  if (i < NXPAD * 256) {
    int r = i >> 8, c = i & 255;
    WxH[i] = (_Float16)((r < 40) ? W_x[r * 256 + c] : 0.0f);
  }
}

// ---------------------------------------------------------------------------
// K1: LayerNorm over C=128, one wave per token, output f16
// ---------------------------------------------------------------------------
__global__ void k_layernorm(const float* __restrict__ x,
                            const float* __restrict__ w,
                            const float* __restrict__ bias,
                            _Float16* __restrict__ xn) {
  int token = (blockIdx.x * blockDim.x + threadIdx.x) >> 5;
  int lane = threadIdx.x & 31;
  int b = token >> 13;
  int l = token & (LSEQ - 1);
  const float* xb = x + (size_t)b * DMODEL * LSEQ + l;
  float v[4];
  float s = 0.f;
#pragma unroll
  for (int j = 0; j < 4; ++j) {
    int c = lane * 4 + j;
    v[j] = xb[(size_t)c * LSEQ];
    s += v[j];
  }
#pragma unroll
  for (int off = 16; off > 0; off >>= 1) s += __shfl_xor(s, off, 32);
  float mu = s * (1.0f / 128.0f);
  float q = 0.f;
#pragma unroll
  for (int j = 0; j < 4; ++j) { float d = v[j] - mu; q += d * d; }
#pragma unroll
  for (int off = 16; off > 0; off >>= 1) q += __shfl_xor(q, off, 32);
  float inv = rsqrtf(q * (1.0f / 128.0f) + 1e-5f);
  _Float16* orow = xn + (size_t)token * DMODEL;
#pragma unroll
  for (int j = 0; j < 4; ++j) {
    int c = lane * 4 + j;
    orow[c] = (_Float16)((v[j] - mu) * inv * w[c] + bias[c]);
  }
}

// ---------------------------------------------------------------------------
// K2: WMMA GEMM  xz[m,e] = sum_c xn[m,c]*W_in[e,c]  (M=16384,K=128,N=512)
// 8 waves/block share one A tile, async-staged into dynamic LDS (4 KB).
// grid: (1024 m-tiles)*(4 n-groups); waves cover nt = ngrp*8 + waveid
// ---------------------------------------------------------------------------
__global__ void k_gemm_xz(const _Float16* __restrict__ A,
                          const _Float16* __restrict__ Bw,
                          float* __restrict__ xi, float* __restrict__ zg) {
  extern __shared__ _Float16 shA[];        // dynamic LDS: 16*DMODEL halfs
  const int tid = threadIdx.x;
  const int mt = blockIdx.x >> 2;
  const int nt = ((blockIdx.x & 3) << 3) + (tid >> 5);
  const int m0 = mt * 16, n0 = nt * 16;

  // stage A tile: 2048 halfs = 256 threads x 16B (dynamic LDS base offset 0)
  async_lds_b128((unsigned)(tid * 16),
                 (const void*)(A + (size_t)m0 * DMODEL + tid * 8));
  async_wait0();
  __syncthreads();

  v8f acc = {};
#pragma unroll
  for (int k0 = 0; k0 < DMODEL; k0 += 32) {
    v16h a = load_fragA_sh(shA, DMODEL, k0);
    v16h b = load_fragB(Bw, DMODEL, n0, k0);
    acc = __builtin_amdgcn_wmma_f32_16x16x32_f16(false, a, false, b,
                                                 (short)0, acc, false, false);
  }
  const int lane = tid & 31;
  const int hi = lane >> 4;
  float* dst = (nt < 16) ? xi : zg;          // uniform per wave
  const int col = ((nt & 15) << 4) + (lane & 15);
#pragma unroll
  for (int r = 0; r < 8; ++r)
    dst[(size_t)(m0 + r + hi * 8) * DINNER + col] = acc[r];
}

// ---------------------------------------------------------------------------
// K3: causal depthwise conv (width 4) + SiLU ; outputs f32 + f16 copies
// ---------------------------------------------------------------------------
__global__ void k_conv_silu(const float* __restrict__ xi,
                            const float* __restrict__ cw,
                            const float* __restrict__ cb,
                            float* __restrict__ xc, _Float16* __restrict__ xch) {
  size_t idx = (size_t)blockIdx.x * blockDim.x + threadIdx.x;
  int d = (int)(idx & 255);
  int l = (int)((idx >> 8) & (LSEQ - 1));
  int b = (int)(idx >> 21);
  float acc = cb[d];
#pragma unroll
  for (int j = 0; j < 4; ++j) {
    int ls = l - 3 + j;
    if (ls >= 0)
      acc += xi[((size_t)(b * LSEQ + ls)) * DINNER + d] * cw[d * 4 + j];
  }
  float s = silu_f(acc);
  xc[idx] = s;
  xch[idx] = (_Float16)s;
}

// ---------------------------------------------------------------------------
// K4a: WMMA GEMM dbl[m,e] = sum_d xc[m,d]*W_x[e,d] (M=16384,K=256,N=48 padded)
// packed store dbl[token][48]: [0:8)=dt_raw, [8:24)=B, [24:40)=C
// ---------------------------------------------------------------------------
__global__ void k_gemm_dbl(const _Float16* __restrict__ A,
                           const _Float16* __restrict__ Bw,
                           float* __restrict__ dbl) {
  int wave = (blockIdx.x << 3) + (threadIdx.x >> 5);
  int mt = wave / 3;
  int nt = wave - mt * 3;
  int m0 = mt * 16, n0 = nt * 16;
  v8f acc = {};
#pragma unroll
  for (int k0 = 0; k0 < DINNER; k0 += 32) {
    v16h a = load_fragA(A, DINNER, m0, k0);
    v16h b = load_fragB(Bw, DINNER, n0, k0);
    acc = __builtin_amdgcn_wmma_f32_16x16x32_f16(false, a, false, b,
                                                 (short)0, acc, false, false);
  }
  int lane = threadIdx.x & 31;
  int col = n0 + (lane & 15);
  int hi = lane >> 4;
#pragma unroll
  for (int r = 0; r < 8; ++r)
    dbl[(size_t)(m0 + r + hi * 8) * NXPAD + col] = acc[r];
}

// ---------------------------------------------------------------------------
// K4b: dt = softplus(dt_raw @ W_dt^T + b_dt)   (K=8, tiny)
// ---------------------------------------------------------------------------
__global__ void k_dt(const float* __restrict__ dbl, const float* __restrict__ Wdt,
                     const float* __restrict__ bdt, float* __restrict__ dts) {
  size_t idx = (size_t)blockIdx.x * blockDim.x + threadIdx.x;
  int d = (int)(idx & 255);
  size_t token = idx >> 8;
  float s = bdt[d];
#pragma unroll
  for (int r = 0; r < DTRANK; ++r)
    s += dbl[token * NXPAD + r] * Wdt[d * DTRANK + r];
  dts[idx] = (s > 20.0f) ? s : log1pf(__expf(s));
}

// ---------------------------------------------------------------------------
// K5: chunk-parallel diagonal linear scan (3 passes).
// block=256: 16 d-channels per block; each 16-lane group = 16 states of one d.
// ---------------------------------------------------------------------------
__global__ void k_scan_chunks(const float* __restrict__ dts,
                              const float* __restrict__ xc,
                              const float* __restrict__ dbl,
                              const float* __restrict__ A_log,
                              float* __restrict__ Hc, float* __restrict__ Pc) {
  int bid = blockIdx.x;                    // b*1024 + chunk*16 + dgrp
  int dgrp = bid & 15;
  int chunk = (bid >> 4) & (NCHUNK - 1);
  int b = bid >> 10;
  int d = dgrp * 16 + (threadIdx.x >> 4);
  int n = threadIdx.x & 15;
  float a = -__expf(A_log[d * DSTATE + n]);
  float h = 0.f, sdt = 0.f;
  size_t t0 = (size_t)b * LSEQ + (size_t)chunk * CHLEN;
  for (int i = 0; i < CHLEN; ++i) {
    size_t t = t0 + i;
    float dtv = dts[t * DINNER + d];
    float xv  = xc[t * DINNER + d];
    float Bv  = dbl[t * NXPAD + DTRANK + n];
    h = __expf(dtv * a) * h + Bv * dtv * xv;
    sdt += dtv;
  }
  size_t idx = (((size_t)b * NCHUNK + chunk) * DINNER + d) * DSTATE + n;
  Hc[idx] = h;
  Pc[idx] = __expf(a * sdt);   // product of per-step decays over the chunk
}

__global__ void k_scan_carry(const float* __restrict__ Hc,
                             const float* __restrict__ Pc,
                             float* __restrict__ carry) {
  int dgrp = blockIdx.x & 15;
  int b = blockIdx.x >> 4;
  int d = dgrp * 16 + (threadIdx.x >> 4);
  int n = threadIdx.x & 15;
  float h = 0.f;
  for (int c = 0; c < NCHUNK; ++c) {
    size_t idx = (((size_t)b * NCHUNK + c) * DINNER + d) * DSTATE + n;
    carry[idx] = h;                        // state entering chunk c
    h = Pc[idx] * h + Hc[idx];
  }
}

__global__ void k_scan_apply(const float* __restrict__ dts,
                             const float* __restrict__ xc,
                             const float* __restrict__ dbl,
                             const float* __restrict__ A_log,
                             const float* __restrict__ carry,
                             float* __restrict__ y) {
  int bid = blockIdx.x;
  int dgrp = bid & 15;
  int chunk = (bid >> 4) & (NCHUNK - 1);
  int b = bid >> 10;
  int d = dgrp * 16 + (threadIdx.x >> 4);
  int n = threadIdx.x & 15;
  float a = -__expf(A_log[d * DSTATE + n]);
  size_t cidx = (((size_t)b * NCHUNK + chunk) * DINNER + d) * DSTATE + n;
  float h = carry[cidx];
  size_t t0 = (size_t)b * LSEQ + (size_t)chunk * CHLEN;
  for (int i = 0; i < CHLEN; ++i) {
    size_t t = t0 + i;
    float dtv = dts[t * DINNER + d];
    float xv  = xc[t * DINNER + d];
    float Bv  = dbl[t * NXPAD + DTRANK + n];
    float Cv  = dbl[t * NXPAD + DTRANK + DSTATE + n];
    h = __expf(dtv * a) * h + Bv * dtv * xv;
    float yc = h * Cv;
    // reduce over 16 states (stays within each 16-lane half of the wave)
    yc += __shfl_xor(yc, 1, 32);
    yc += __shfl_xor(yc, 2, 32);
    yc += __shfl_xor(yc, 4, 32);
    yc += __shfl_xor(yc, 8, 32);
    if (n == 0) y[t * DINNER + d] = yc;
  }
}

// ---------------------------------------------------------------------------
// K6: y = (y + xc*D) * silu(z), stored f16 for the output GEMM
// ---------------------------------------------------------------------------
__global__ void k_gate(const float* __restrict__ y, const float* __restrict__ xc,
                       const float* __restrict__ zg, const float* __restrict__ Dv,
                       _Float16* __restrict__ yh) {
  size_t idx = (size_t)blockIdx.x * blockDim.x + threadIdx.x;
  int d = (int)(idx & 255);
  float v = (y[idx] + xc[idx] * Dv[d]) * silu_f(zg[idx]);
  yh[idx] = (_Float16)v;
}

// ---------------------------------------------------------------------------
// K7: WMMA GEMM  out[m,c] = sum_d y[m,d]*W_out[c,d]  (M=16384,K=256,N=128)
// 8 waves/block share one A tile, async-staged into dynamic LDS (8 KB).
// ---------------------------------------------------------------------------
__global__ void k_gemm_out(const _Float16* __restrict__ A,
                           const _Float16* __restrict__ Bw,
                           float* __restrict__ out) {
  extern __shared__ _Float16 shA[];        // dynamic LDS: 16*DINNER halfs
  const int tid = threadIdx.x;
  const int mt = blockIdx.x;
  const int nt = tid >> 5;                 // 8 n-tiles, one per wave
  const int m0 = mt * 16, n0 = nt * 16;

  // stage A tile: 4096 halfs = 256 threads x 2 x 16B
  const _Float16* src = A + (size_t)m0 * DINNER + tid * 8;
  async_lds_b128((unsigned)(tid * 16), (const void*)src);
  async_lds_b128((unsigned)(4096 + tid * 16), (const void*)(src + 2048));
  async_wait0();
  __syncthreads();

  v8f acc = {};
#pragma unroll
  for (int k0 = 0; k0 < DINNER; k0 += 32) {
    v16h a = load_fragA_sh(shA, DINNER, k0);
    v16h b = load_fragB(Bw, DINNER, n0, k0);
    acc = __builtin_amdgcn_wmma_f32_16x16x32_f16(false, a, false, b,
                                                 (short)0, acc, false, false);
  }
  const int lane = tid & 31;
  const int col = n0 + (lane & 15);
  const int hi = lane >> 4;
#pragma unroll
  for (int r = 0; r < 8; ++r) {
    int row = m0 + r + hi * 8;
    int b = row >> 13;
    int l = row & (LSEQ - 1);
    out[(size_t)b * DMODEL * LSEQ + (size_t)col * LSEQ + l] = acc[r];
  }
}

// ---------------------------------------------------------------------------
extern "C" void kernel_launch(void* const* d_in, const int* in_sizes, int n_in,
                              void* d_out, int out_size, void* d_ws, size_t ws_size,
                              hipStream_t stream) {
  (void)in_sizes; (void)n_in; (void)out_size; (void)ws_size;
  const float* x      = (const float*)d_in[0];
  const float* ln_w   = (const float*)d_in[1];
  const float* ln_b   = (const float*)d_in[2];
  const float* W_in   = (const float*)d_in[3];
  const float* conv_w = (const float*)d_in[4];
  const float* conv_b = (const float*)d_in[5];
  const float* W_x    = (const float*)d_in[6];
  const float* W_dt   = (const float*)d_in[7];
  const float* b_dt   = (const float*)d_in[8];
  const float* A_log  = (const float*)d_in[9];
  const float* Dv     = (const float*)d_in[10];
  const float* W_out  = (const float*)d_in[11];
  float* out = (float*)d_out;

  // workspace carve-up (256B aligned)
  char* w = (char*)d_ws;
  size_t off = 0;
  auto carve = [&](size_t bytes) -> void* {
    void* p = w + off;
    off += (bytes + 255) & ~(size_t)255;
    return p;
  };
  _Float16* xnH   = (_Float16*)carve((size_t)NTOK * DMODEL * 2);
  _Float16* WinH  = (_Float16*)carve((size_t)512 * 128 * 2);
  _Float16* WoutH = (_Float16*)carve((size_t)128 * 256 * 2);
  _Float16* WxH   = (_Float16*)carve((size_t)NXPAD * 256 * 2);
  float*    xi    = (float*)carve((size_t)NTOK * DINNER * 4);
  float*    zg    = (float*)carve((size_t)NTOK * DINNER * 4);
  float*    xc    = (float*)carve((size_t)NTOK * DINNER * 4);
  _Float16* xcH   = (_Float16*)carve((size_t)NTOK * DINNER * 2);
  float*    dbl   = (float*)carve((size_t)NTOK * NXPAD * 4);
  float*    dts   = (float*)carve((size_t)NTOK * DINNER * 4);
  float*    Hc    = (float*)carve((size_t)BSZ * NCHUNK * DINNER * DSTATE * 4);
  float*    Pc    = (float*)carve((size_t)BSZ * NCHUNK * DINNER * DSTATE * 4);
  float*    cry   = (float*)carve((size_t)BSZ * NCHUNK * DINNER * DSTATE * 4);
  float*    yv    = (float*)carve((size_t)NTOK * DINNER * 4);
  _Float16* yH    = (_Float16*)carve((size_t)NTOK * DINNER * 2);

  // K0: weight conversion (f32 -> f16, W_x zero-padded to 48 rows)
  k_cvt_weights<<<(512 * 128 + 255) / 256, 256, 0, stream>>>(
      W_in, W_out, W_x, WinH, WoutH, WxH);
  // K1: layernorm (1 wave/token, 8 tokens/block)
  k_layernorm<<<NTOK / 8, 256, 0, stream>>>(x, ln_w, ln_b, xnH);
  // K2: input projection (WMMA + async-LDS A tile, 4 KB dynamic LDS)
  k_gemm_xz<<<1024 * 4, 256, 16 * DMODEL * 2, stream>>>(xnH, WinH, xi, zg);
  // K3: depthwise causal conv + SiLU
  k_conv_silu<<<(NTOK * DINNER) / 256, 256, 0, stream>>>(xi, conv_w, conv_b, xc, xcH);
  // K4: x-projection as WMMA (N=48 padded), then dt softplus
  k_gemm_dbl<<<(1024 * 3) / 8, 256, 0, stream>>>(xcH, WxH, dbl);
  k_dt<<<(NTOK * DINNER) / 256, 256, 0, stream>>>(dbl, W_dt, b_dt, dts);
  // K5: chunk-parallel selective scan
  k_scan_chunks<<<BSZ * NCHUNK * 16, 256, 0, stream>>>(dts, xc, dbl, A_log, Hc, Pc);
  k_scan_carry<<<BSZ * 16, 256, 0, stream>>>(Hc, Pc, cry);
  k_scan_apply<<<BSZ * NCHUNK * 16, 256, 0, stream>>>(dts, xc, dbl, A_log, cry, yv);
  // K6: skip + gate, produce f16 operand
  k_gate<<<(NTOK * DINNER) / 256, 256, 0, stream>>>(yv, xc, zg, Dv, yH);
  // K7: output projection (WMMA + async-LDS A tile, 8 KB dynamic LDS)
  k_gemm_out<<<1024, 256, 16 * DINNER * 2, stream>>>(yH, WoutH, out);
}
